// HybridPointNet_68281390072582
// MI455X (gfx1250) — compile-verified, matrix operand order checked by reading, not run on hardware
//
#include <hip/hip_runtime.h>
#include <math.h>

// ---- problem constants (match reference) ----
#define Bc       2
#define Nc       8192
#define Sc       2048     // NPOINT == attention sequence length
#define NSAMPLEc 32
#define C1c      128
#define Cc       256
#define Hc       8
#define HDc      32
#define R2c      0.04f    // RADIUS^2

typedef __bf16 bf16;
typedef __attribute__((ext_vector_type(16))) __bf16 v16bf;
typedef __attribute__((ext_vector_type(8)))  __bf16 v8bf;
typedef __attribute__((ext_vector_type(8)))  float  v8f;

__device__ __forceinline__ bf16 f2bf(float f) {
  unsigned u  = __builtin_bit_cast(unsigned int, f);
  unsigned rb = ((u >> 16) & 1u) + 0x7fffu;   // round-to-nearest-even
  u += rb;
  unsigned short h = (unsigned short)(u >> 16);
  return __builtin_bit_cast(__bf16, h);
}

// Load a 16-bit WMMA fragment (A layout, or B layout via row==column of W^T).
// rowptr points at element k-base of the row; lane-half selects K sub-ranges:
//   half h: elements 0..7  <- K = h*8 .. h*8+7        (contiguous, 16B)
//           elements 8..15 <- K = 16+h*8 .. 16+h*8+7  (contiguous, 16B)
__device__ __forceinline__ v16bf load_frag(const bf16* rowptr, int lane) {
  int half = (lane >> 4) & 1;
  const bf16* p = rowptr + half * 8;
  v8bf lo = *(const v8bf*)(p);
  v8bf hi = *(const v8bf*)(p + 16);
  v16bf f;
#pragma unroll
  for (int i = 0; i < 8; ++i) { f[i] = lo[i]; f[8 + i] = hi[i]; }
  return f;
}

__device__ __forceinline__ v8f wmma_bf16(v16bf a, v16bf b, v8f c) {
  return __builtin_amdgcn_wmma_f32_16x16x32_bf16(false, a, false, b, (short)0, c, false, false);
}

// ------------------------------------------------------------------
// 1) Farthest point sampling: one block per batch, dist in registers
// ------------------------------------------------------------------
__global__ __launch_bounds__(256)
void fps_kernel(const float* __restrict__ xyz, float* __restrict__ newxyz) {
  int b = blockIdx.x;
  int t = threadIdx.x;
  const float* X = xyz + (size_t)b * Nc * 3;
  float dloc[32];
#pragma unroll
  for (int j = 0; j < 32; ++j) dloc[j] = 1e10f;
  __shared__ float redv[256];
  __shared__ int   redi[256];
  __shared__ int   sfar;
  if (t == 0) sfar = 0;
  __syncthreads();
  for (int it = 0; it < Sc; ++it) {
    int far = sfar;
    float cx = X[far * 3 + 0], cy = X[far * 3 + 1], cz = X[far * 3 + 2];
    if (t == 0) {
      float* o = newxyz + ((size_t)b * Sc + it) * 3;
      o[0] = cx; o[1] = cy; o[2] = cz;
    }
    float bm = -1.f; int bi = t * 32;
#pragma unroll
    for (int j = 0; j < 32; ++j) {
      int i = t * 32 + j;
      float dx = X[i * 3 + 0] - cx, dy = X[i * 3 + 1] - cy, dz = X[i * 3 + 2] - cz;
      float d  = dx * dx + dy * dy + dz * dz;
      float nd = fminf(dloc[j], d);
      dloc[j]  = nd;
      if (nd > bm) { bm = nd; bi = i; }
    }
    redv[t] = bm; redi[t] = bi;
    __syncthreads();
    for (int s = 128; s > 0; s >>= 1) {
      if (t < s) {
        float ov = redv[t + s]; int oi = redi[t + s];
        if (ov > redv[t] || (ov == redv[t] && oi < redi[t])) { redv[t] = ov; redi[t] = oi; }
      }
      __syncthreads();
    }
    if (t == 0) sfar = redi[0];
    __syncthreads();
  }
}

// ------------------------------------------------------------------
// 2) Ball query: one wave per center; first 32 in-radius indices
//    in ascending order (== sort + truncate in the reference)
// ------------------------------------------------------------------
__global__ __launch_bounds__(256)
void ballquery_kernel(const float* __restrict__ xyz, const float* __restrict__ newxyz,
                      int* __restrict__ gidx) {
  int wid  = (blockIdx.x * blockDim.x + threadIdx.x) >> 5;
  int lane = threadIdx.x & 31;
  int b = wid / Sc, s = wid % Sc;
  const float* c = newxyz + ((size_t)b * Sc + s) * 3;
  float cx = c[0], cy = c[1], cz = c[2];
  const float* X = xyz + (size_t)b * Nc * 3;
  int cnt = 0, myidx = 0, first = 0;
  for (int base = 0; base < Nc && cnt < NSAMPLEc; base += 32) {
    int i = base + lane;
    float dx = X[i * 3 + 0] - cx, dy = X[i * 3 + 1] - cy, dz = X[i * 3 + 2] - cz;
    float d2 = dx * dx + dy * dy + dz * dz;
    unsigned long long mask = __ballot(d2 <= R2c);
    while (mask && cnt < NSAMPLEc) {
      int bit = __ffsll((unsigned long long)mask) - 1;
      int idx = base + bit;
      if (lane == cnt) myidx = idx;
      if (cnt == 0) first = idx;
      ++cnt;
      mask &= (mask - 1);
    }
  }
  if (lane >= cnt) myidx = first;
  gidx[((size_t)b * Sc + s) * NSAMPLEc + lane] = myidx;
}

// ------------------------------------------------------------------
// 3) Grouped max-pool of both feature sets -> bf16 GEMM inputs
// ------------------------------------------------------------------
__global__ __launch_bounds__(128)
void groupmax_kernel(const float* __restrict__ f1, const float* __restrict__ f2,
                     const int* __restrict__ gidx,
                     bf16* __restrict__ o1, bf16* __restrict__ o2) {
  int bs = blockIdx.x;            // 0 .. B*Sc
  int c  = threadIdx.x;           // 0 .. 127
  int b  = bs / Sc;
  const int* gi = gidx + (size_t)bs * NSAMPLEc;
  float m1 = -3.4e38f, m2 = -3.4e38f;
#pragma unroll 4
  for (int j = 0; j < NSAMPLEc; ++j) {
    int idx = gi[j];
    size_t r = ((size_t)b * Nc + idx) * C1c + c;
    m1 = fmaxf(m1, f1[r]);
    m2 = fmaxf(m2, f2[r]);
  }
  o1[(size_t)bs * C1c + c] = f2bf(m1);
  o2[(size_t)bs * C1c + c] = f2bf(m2);
}

// ------------------------------------------------------------------
// 4) Weight transpose+convert: W (K x N) f32 -> Wt (N x K) bf16
// ------------------------------------------------------------------
__global__ __launch_bounds__(256)
void wtrans_kernel(const float* __restrict__ W, bf16* __restrict__ Wt, int K, int Nout) {
  int e = blockIdx.x * blockDim.x + threadIdx.x;
  if (e >= K * Nout) return;
  int k = e / Nout, n = e % Nout;
  Wt[(size_t)n * K + k] = f2bf(W[e]);
}

// ------------------------------------------------------------------
// 5) WMMA GEMM: out = X(MxK) @ Wt^T + bias.
//    One wave computes a 32x64 output tile: per k-step, 2 A-frags +
//    4 B-frags feed 8 WMMAs (2.7x better load:FLOP ratio than 16x16).
//    act: 0=none, 1=sigmoid.  outf/outbf optional.
// ------------------------------------------------------------------
__global__ __launch_bounds__(128)
void wmma_gemm_kernel(const bf16* __restrict__ X, const bf16* __restrict__ Wt,
                      const float* __restrict__ bias,
                      float* __restrict__ outf, bf16* __restrict__ outbf,
                      int M, int Nout, int K, int act) {
  int wave = blockIdx.x * (blockDim.x >> 5) + (threadIdx.x >> 5);
  int lane = threadIdx.x & 31;
  int tilesN = Nout >> 6;                       // 64-wide tiles
  if (wave >= (M >> 5) * tilesN) return;        // wave-uniform: EXEC stays full
  int tm = wave / tilesN, tn = wave % tilesN;
  int row = lane & 15;

  const bf16* Xr0 = X  + (size_t)(tm * 32 + row)      * K;
  const bf16* Xr1 = X  + (size_t)(tm * 32 + 16 + row) * K;
  const bf16* Wr0 = Wt + (size_t)(tn * 64 + row)      * K;
  const bf16* Wr1 = Wt + (size_t)(tn * 64 + 16 + row) * K;
  const bf16* Wr2 = Wt + (size_t)(tn * 64 + 32 + row) * K;
  const bf16* Wr3 = Wt + (size_t)(tn * 64 + 48 + row) * K;

  v8f zero = {};
  v8f acc[8];                                   // [mi*4 + j]
#pragma unroll
  for (int i = 0; i < 8; ++i) acc[i] = zero;

  for (int k = 0; k < K; k += 32) {
    __builtin_prefetch(Xr0 + k + 64, 0, 1);
    v16bf a0 = load_frag(Xr0 + k, lane);
    v16bf a1 = load_frag(Xr1 + k, lane);
    v16bf w0 = load_frag(Wr0 + k, lane);
    v16bf w1 = load_frag(Wr1 + k, lane);
    v16bf w2 = load_frag(Wr2 + k, lane);
    v16bf w3 = load_frag(Wr3 + k, lane);
    acc[0] = wmma_bf16(a0, w0, acc[0]);
    acc[1] = wmma_bf16(a0, w1, acc[1]);
    acc[2] = wmma_bf16(a0, w2, acc[2]);
    acc[3] = wmma_bf16(a0, w3, acc[3]);
    acc[4] = wmma_bf16(a1, w0, acc[4]);
    acc[5] = wmma_bf16(a1, w1, acc[5]);
    acc[6] = wmma_bf16(a1, w2, acc[6]);
    acc[7] = wmma_bf16(a1, w3, acc[7]);
  }

#pragma unroll
  for (int mi = 0; mi < 2; ++mi) {
    int r0 = tm * 32 + mi * 16 + ((lane >> 4) << 3);
#pragma unroll
    for (int j = 0; j < 4; ++j) {
      int col = tn * 64 + j * 16 + row;
      float bv = bias ? bias[col] : 0.f;
      v8f a = acc[mi * 4 + j];
#pragma unroll
      for (int i = 0; i < 8; ++i) {
        float v = a[i] + bv;
        if (act == 1) v = 1.f / (1.f + __expf(-v));
        size_t o = (size_t)(r0 + i) * Nout + col;
        if (outf)  outf[o]  = v;
        if (outbf) outbf[o] = f2bf(v);
      }
    }
  }
}

// ------------------------------------------------------------------
// 6) V transpose: (B,S,C) bf16 -> (B,H,HD,S) bf16
// ------------------------------------------------------------------
__global__ __launch_bounds__(256)
void vtrans_kernel(const bf16* __restrict__ v, bf16* __restrict__ vt) {
  int i = blockIdx.x * blockDim.x + threadIdx.x;       // over B*S*C
  if (i >= Bc * Sc * Cc) return;
  int b = i / (Sc * Cc);
  int rem = i % (Sc * Cc);
  int s = rem / Cc, c = rem % Cc;
  int h = c >> 5, d = c & 31;
  vt[(((size_t)b * Hc + h) * HDc + d) * Sc + s] = v[i];
}

// ------------------------------------------------------------------
// 7) Flash attention, 1 wave per (b,h,32-query block) = two 16-query
//    tiles sharing every K/V fragment (8 WMMAs per 4 fragment loads).
//    Computes S^T = K @ Q^T so the per-query softmax is lane-local and
//    the exp'd score registers are already in B-fragment layout for
//    the P.V WMMA (zero cross-lane movement).
// ------------------------------------------------------------------
__global__ __launch_bounds__(128)
void flash_attn_kernel(const bf16* __restrict__ Qb, const bf16* __restrict__ Kb,
                       const bf16* __restrict__ Vt, float* __restrict__ out) {
  int wave = blockIdx.x * (blockDim.x >> 5) + (threadIdx.x >> 5);
  int lane = threadIdx.x & 31;
  const int qblocks = Sc / 32;
  if (wave >= Bc * Hc * qblocks) return;        // wave-uniform
  int qt = wave % qblocks;
  int bh = wave / qblocks;
  int h = bh % Hc, b = bh / Hc;

  const bf16* Qbase = Qb + (size_t)b * Sc * Cc + h * HDc;
  const bf16* Kbase = Kb + (size_t)b * Sc * Cc + h * HDc;
  const bf16* Vbase = Vt + ((size_t)(b * Hc + h) * HDc) * Sc;

  int r = lane & 15;
  v16bf qf[2];
  qf[0] = load_frag(Qbase + (size_t)(qt * 32 + r)      * Cc, lane);
  qf[1] = load_frag(Qbase + (size_t)(qt * 32 + 16 + r) * Cc, lane);

  v8f zero = {};
  v8f ot0[2], ot1[2];
  ot0[0] = zero; ot0[1] = zero; ot1[0] = zero; ot1[1] = zero;
  float m[2]    = { -1e30f, -1e30f };
  float lsum[2] = { 0.f, 0.f };
  const float sc = 0.17677669529663687f;        // 1/sqrt(HD)

  for (int kc = 0; kc < Sc; kc += 32) {
    v16bf kf0 = load_frag(Kbase + (size_t)(kc + r)      * Cc, lane);
    v16bf kf1 = load_frag(Kbase + (size_t)(kc + 16 + r) * Cc, lane);

    v16bf pf[2];
#pragma unroll
    for (int t = 0; t < 2; ++t) {
      v8f s0 = wmma_bf16(kf0, qf[t], zero);     // keys kc..kc+15  x 16 queries
      v8f s1 = wmma_bf16(kf1, qf[t], zero);     // keys kc+16..+31 x 16 queries

      float lm = -1e30f;
#pragma unroll
      for (int i = 0; i < 8; ++i) {
        s0[i] *= sc; s1[i] *= sc;
        lm = fmaxf(lm, fmaxf(s0[i], s1[i]));
      }
      lm = fmaxf(lm, __shfl_xor(lm, 16, 32));   // partner half: same query, other keys
      float mnew = fmaxf(m[t], lm);

      float p0[8], p1[8], ls = 0.f;
#pragma unroll
      for (int i = 0; i < 8; ++i) {
        p0[i] = __expf(s0[i] - mnew);
        p1[i] = __expf(s1[i] - mnew);
        ls += p0[i] + p1[i];
      }
      ls += __shfl_xor(ls, 16, 32);
      float alpha = __expf(m[t] - mnew);
      lsum[t] = lsum[t] * alpha + ls;
      m[t] = mnew;
#pragma unroll
      for (int i = 0; i < 8; ++i) { ot0[t][i] *= alpha; ot1[t][i] *= alpha; }

      // D-tile registers map 1:1 (lane-local) onto the B-fragment layout
#pragma unroll
      for (int i = 0; i < 8; ++i) { pf[t][i] = f2bf(p0[i]); pf[t][8 + i] = f2bf(p1[i]); }
    }

    v16bf vf0 = load_frag(Vbase + (size_t)r        * Sc + kc, lane);  // dims 0..15
    v16bf vf1 = load_frag(Vbase + (size_t)(16 + r) * Sc + kc, lane);  // dims 16..31
#pragma unroll
    for (int t = 0; t < 2; ++t) {
      ot0[t] = wmma_bf16(vf0, pf[t], ot0[t]);
      ot1[t] = wmma_bf16(vf1, pf[t], ot1[t]);
    }
  }

#pragma unroll
  for (int t = 0; t < 2; ++t) {
    float inv = (lsum[t] > 0.f) ? 1.f / lsum[t] : 0.f;
    int q  = qt * 32 + t * 16 + r;
    int d0 = (lane >> 4) << 3;
    float* orow = out + ((size_t)b * Sc + q) * Cc + h * HDc;
#pragma unroll
    for (int i = 0; i < 8; ++i) {
      orow[d0 + i]      = ot0[t][i] * inv;
      orow[16 + d0 + i] = ot1[t][i] * inv;
    }
  }
}

// ------------------------------------------------------------------
// 8) Elementwise glue kernels
// ------------------------------------------------------------------
__global__ __launch_bounds__(256)
void build_gatein_kernel(const float* __restrict__ o1, const float* __restrict__ o2,
                         const float* __restrict__ pn2f, const float* __restrict__ ptf,
                         bf16* __restrict__ gi) {
  int i = blockIdx.x * blockDim.x + threadIdx.x;
  if (i >= Bc * Sc * Cc) return;
  int row = i / Cc, c = i % Cc;
  size_t rbase = (size_t)row * (3 * Cc);
  gi[rbase + c]           = f2bf(o1[i]);
  gi[rbase + Cc + c]      = f2bf(o2[i]);
  gi[rbase + 2 * Cc + c]  = f2bf(pn2f[i] + ptf[i]);
}

__global__ __launch_bounds__(256)
void build_fused_kernel(const float* __restrict__ o1, const float* __restrict__ o2,
                        const float* __restrict__ gate, bf16* __restrict__ fu) {
  int i = blockIdx.x * blockDim.x + threadIdx.x;
  if (i >= Bc * Sc * Cc) return;
  int row = i / Cc, c = i % Cc;
  float g = gate[i];
  size_t rbase = (size_t)row * (2 * Cc);
  fu[rbase + c]      = f2bf(o1[i] * g);
  fu[rbase + Cc + c] = f2bf(o2[i] * (1.f - g));
}

__global__ __launch_bounds__(256)
void ln_gelu_kernel(const float* __restrict__ x, const float* __restrict__ g,
                    const float* __restrict__ bta, bf16* __restrict__ o) {
  int row = blockIdx.x, t = threadIdx.x;
  __shared__ float red[256];
  float v = x[(size_t)row * Cc + t];
  red[t] = v; __syncthreads();
  for (int s = 128; s > 0; s >>= 1) { if (t < s) red[t] += red[t + s]; __syncthreads(); }
  float mean = red[0] * (1.f / Cc); __syncthreads();
  float dv = v - mean;
  red[t] = dv * dv; __syncthreads();
  for (int s = 128; s > 0; s >>= 1) { if (t < s) red[t] += red[t + s]; __syncthreads(); }
  float var = red[0] * (1.f / Cc);
  float y = dv * rsqrtf(var + 1e-5f) * g[t] + bta[t];
  float ge = 0.5f * y * (1.f + erff(y * 0.70710678118654752f));   // exact GELU
  o[(size_t)row * Cc + t] = f2bf(ge);
}

__global__ __launch_bounds__(256)
void final_ln_kernel(const float* __restrict__ h2, const float* __restrict__ pn2f,
                     const float* __restrict__ ptf, const float* __restrict__ g,
                     const float* __restrict__ bta, float* __restrict__ out) {
  int row = blockIdx.x, t = threadIdx.x;
  __shared__ float red[256];
  size_t i = (size_t)row * Cc + t;
  float v = h2[i] + pn2f[i] + ptf[i];
  red[t] = v; __syncthreads();
  for (int s = 128; s > 0; s >>= 1) { if (t < s) red[t] += red[t + s]; __syncthreads(); }
  float mean = red[0] * (1.f / Cc); __syncthreads();
  float dv = v - mean;
  red[t] = dv * dv; __syncthreads();
  for (int s = 128; s > 0; s >>= 1) { if (t < s) red[t] += red[t + s]; __syncthreads(); }
  float var = red[0] * (1.f / Cc);
  out[i] = dv * rsqrtf(var + 1e-5f) * g[t] + bta[t];
}

// ------------------------------------------------------------------
// Launcher
// ------------------------------------------------------------------
extern "C" void kernel_launch(void* const* d_in, const int* in_sizes, int n_in,
                              void* d_out, int out_size, void* d_ws, size_t ws_size,
                              hipStream_t stream) {
  (void)in_sizes; (void)n_in; (void)out_size; (void)ws_size;

  const float* xyz   = (const float*)d_in[0];
  const float* pn2ft = (const float*)d_in[1];
  const float* ptft  = (const float*)d_in[2];
  const float* Wpn2  = (const float*)d_in[3];  const float* bpn2 = (const float*)d_in[4];
  const float* Wpt   = (const float*)d_in[5];  const float* bpt  = (const float*)d_in[6];
  const float* Wq1   = (const float*)d_in[7];  const float* bq1  = (const float*)d_in[8];
  const float* Wk1   = (const float*)d_in[9];  const float* bk1  = (const float*)d_in[10];
  const float* Wv1   = (const float*)d_in[11]; const float* bv1  = (const float*)d_in[12];
  const float* Wq2   = (const float*)d_in[13]; const float* bq2  = (const float*)d_in[14];
  const float* Wk2   = (const float*)d_in[15]; const float* bk2  = (const float*)d_in[16];
  const float* Wv2   = (const float*)d_in[17]; const float* bv2  = (const float*)d_in[18];
  const float* Wg    = (const float*)d_in[19]; const float* bg   = (const float*)d_in[20];
  const float* Wf1   = (const float*)d_in[21]; const float* bf1  = (const float*)d_in[22];
  const float* Wf2   = (const float*)d_in[23]; const float* bf2  = (const float*)d_in[24];
  const float* g1    = (const float*)d_in[25]; const float* b1   = (const float*)d_in[26];
  const float* gN    = (const float*)d_in[27]; const float* bN   = (const float*)d_in[28];

  float* out = (float*)d_out;

  // ---- workspace bump allocator (256B aligned) ----
  char* ws = (char*)d_ws;
  size_t off = 0;
  auto alloc = [&](size_t bytes) -> void* {
    void* p = ws + off;
    off = (off + bytes + 255) & ~(size_t)255;
    return p;
  };
  const size_t MS = (size_t)Bc * Sc;          // 4096 rows

  float* newxyz   = (float*)alloc(MS * 3 * sizeof(float));
  int*   gidx     = (int*)  alloc(MS * NSAMPLEc * sizeof(int));
  bf16*  pn2lb    = (bf16*) alloc(MS * C1c * sizeof(bf16));
  bf16*  ptlb     = (bf16*) alloc(MS * C1c * sizeof(bf16));
  bf16*  Wpn2t    = (bf16*) alloc((size_t)Cc * C1c * sizeof(bf16));
  bf16*  Wptt     = (bf16*) alloc((size_t)Cc * C1c * sizeof(bf16));
  bf16*  Wq1t     = (bf16*) alloc((size_t)Cc * Cc * sizeof(bf16));
  bf16*  Wk1t     = (bf16*) alloc((size_t)Cc * Cc * sizeof(bf16));
  bf16*  Wv1t     = (bf16*) alloc((size_t)Cc * Cc * sizeof(bf16));
  bf16*  Wq2t     = (bf16*) alloc((size_t)Cc * Cc * sizeof(bf16));
  bf16*  Wk2t     = (bf16*) alloc((size_t)Cc * Cc * sizeof(bf16));
  bf16*  Wv2t     = (bf16*) alloc((size_t)Cc * Cc * sizeof(bf16));
  bf16*  Wgt      = (bf16*) alloc((size_t)Cc * 3 * Cc * sizeof(bf16));
  bf16*  Wf1t     = (bf16*) alloc((size_t)Cc * 2 * Cc * sizeof(bf16));
  bf16*  Wf2t     = (bf16*) alloc((size_t)Cc * Cc * sizeof(bf16));
  float* pn2f     = (float*)alloc(MS * Cc * sizeof(float));
  float* ptf      = (float*)alloc(MS * Cc * sizeof(float));
  bf16*  pn2fb    = (bf16*) alloc(MS * Cc * sizeof(bf16));
  bf16*  ptfb     = (bf16*) alloc(MS * Cc * sizeof(bf16));
  bf16*  q1b      = (bf16*) alloc(MS * Cc * sizeof(bf16));
  bf16*  k1b      = (bf16*) alloc(MS * Cc * sizeof(bf16));
  bf16*  v1b      = (bf16*) alloc(MS * Cc * sizeof(bf16));
  bf16*  q2b      = (bf16*) alloc(MS * Cc * sizeof(bf16));
  bf16*  k2b      = (bf16*) alloc(MS * Cc * sizeof(bf16));
  bf16*  v2b      = (bf16*) alloc(MS * Cc * sizeof(bf16));
  bf16*  vt1b     = (bf16*) alloc(MS * Cc * sizeof(bf16));
  bf16*  vt2b     = (bf16*) alloc(MS * Cc * sizeof(bf16));
  float* out1     = (float*)alloc(MS * Cc * sizeof(float));
  float* out2     = (float*)alloc(MS * Cc * sizeof(float));
  bf16*  gateinb  = (bf16*) alloc(MS * 3 * Cc * sizeof(bf16));
  float* gate     = (float*)alloc(MS * Cc * sizeof(float));
  bf16*  fusedb   = (bf16*) alloc(MS * 2 * Cc * sizeof(bf16));
  float* h1       = (float*)alloc(MS * Cc * sizeof(float));
  bf16*  actb     = (bf16*) alloc(MS * Cc * sizeof(bf16));
  float* h2       = (float*)alloc(MS * Cc * sizeof(float));

  // ---- weight transposes (f32 -> bf16, N x K) ----
  auto wt = [&](const float* W, bf16* Wt, int K, int Nout) {
    int n = K * Nout;
    wtrans_kernel<<<(n + 255) / 256, 256, 0, stream>>>(W, Wt, K, Nout);
  };
  wt(Wpn2, Wpn2t, C1c, Cc);  wt(Wpt, Wptt, C1c, Cc);
  wt(Wq1, Wq1t, Cc, Cc);     wt(Wk1, Wk1t, Cc, Cc);   wt(Wv1, Wv1t, Cc, Cc);
  wt(Wq2, Wq2t, Cc, Cc);     wt(Wk2, Wk2t, Cc, Cc);   wt(Wv2, Wv2t, Cc, Cc);
  wt(Wg,  Wgt,  3 * Cc, Cc); wt(Wf1, Wf1t, 2 * Cc, Cc); wt(Wf2, Wf2t, Cc, Cc);

  // ---- geometry pipeline ----
  fps_kernel<<<Bc, 256, 0, stream>>>(xyz, newxyz);
  ballquery_kernel<<<(int)(MS * 32 / 256), 256, 0, stream>>>(xyz, newxyz, gidx);
  groupmax_kernel<<<(int)MS, 128, 0, stream>>>(pn2ft, ptft, gidx, pn2lb, ptlb);

  // ---- GEMM chain: one wave per 32x64 tile -> (M/32)*(N/64) waves ----
  auto gemm = [&](const bf16* X, const bf16* Wt_, const float* bias,
                  float* of, bf16* ob, int M, int Nout, int K, int act) {
    int tiles = (M >> 5) * (Nout >> 6);        // 512 for all GEMMs here
    wmma_gemm_kernel<<<tiles / 4, 128, 0, stream>>>(X, Wt_, bias, of, ob, M, Nout, K, act);
  };
  gemm(pn2lb, Wpn2t, bpn2, pn2f, pn2fb, (int)MS, Cc, C1c, 0);
  gemm(ptlb,  Wptt,  bpt,  ptf,  ptfb,  (int)MS, Cc, C1c, 0);
  gemm(pn2fb, Wq1t, bq1, nullptr, q1b, (int)MS, Cc, Cc, 0);
  gemm(ptfb,  Wk1t, bk1, nullptr, k1b, (int)MS, Cc, Cc, 0);
  gemm(ptfb,  Wv1t, bv1, nullptr, v1b, (int)MS, Cc, Cc, 0);
  gemm(ptfb,  Wq2t, bq2, nullptr, q2b, (int)MS, Cc, Cc, 0);
  gemm(pn2fb, Wk2t, bk2, nullptr, k2b, (int)MS, Cc, Cc, 0);
  gemm(pn2fb, Wv2t, bv2, nullptr, v2b, (int)MS, Cc, Cc, 0);

  // ---- attention ----
  int nelem = Bc * Sc * Cc;
  vtrans_kernel<<<(nelem + 255) / 256, 256, 0, stream>>>(v1b, vt1b);
  vtrans_kernel<<<(nelem + 255) / 256, 256, 0, stream>>>(v2b, vt2b);
  const int attn_waves = Bc * Hc * (Sc / 32);      // 1024
  flash_attn_kernel<<<attn_waves / 4, 128, 0, stream>>>(q1b, k1b, vt1b, out1);
  flash_attn_kernel<<<attn_waves / 4, 128, 0, stream>>>(q2b, k2b, vt2b, out2);

  // ---- gate + fuse + MLP + layernorms ----
  build_gatein_kernel<<<(nelem + 255) / 256, 256, 0, stream>>>(out1, out2, pn2f, ptf, gateinb);
  gemm(gateinb, Wgt, bg, gate, nullptr, (int)MS, Cc, 3 * Cc, 1);   // fused sigmoid
  build_fused_kernel<<<(nelem + 255) / 256, 256, 0, stream>>>(out1, out2, gate, fusedb);
  gemm(fusedb, Wf1t, bf1, h1, nullptr, (int)MS, Cc, 2 * Cc, 0);
  ln_gelu_kernel<<<(int)MS, 256, 0, stream>>>(h1, g1, b1, actb);
  gemm(actb, Wf2t, bf2, h2, nullptr, (int)MS, Cc, Cc, 0);
  final_ln_kernel<<<(int)MS, 256, 0, stream>>>(h2, pn2f, ptf, gN, bN, out);
}
